// RWKV_Tmix_x060_42915313222103
// MI455X (gfx1250) — compile-verified, matrix operand order checked by reading, not run on hardware
//
#include <hip/hip_runtime.h>
#include <hip/hip_bf16.h>
#include <math.h>

// Problem dims (fixed by reference)
#define BDIM 8
#define TDIM 2048
#define CDIM 512
#define HDIM 8
#define NDIM 64
#define DMIXD 32
#define DDECD 64

typedef __attribute__((ext_vector_type(16))) __bf16 v16bf;
typedef __attribute__((ext_vector_type(8)))  __bf16 v8bf;
typedef __attribute__((ext_vector_type(8)))  float  v8f;

// ---------------------------------------------------------------------------
// bf16 fragment loader (16x32 tile, 16-bit A/B layout per CDNA5 ISA 7.12.2:
// lanes 0-15 hold K={0..7,16..23}, lanes 16-31 hold K={8..15,24..31};
// row = lane & 15). Two 16-byte loads -> global_load_b128, no conversion.
// ---------------------------------------------------------------------------
__device__ __forceinline__ v16bf load_frag(const __bf16* __restrict__ base,
                                           int ldm, int row0, int k0, int lane) {
  const int l  = lane & 15;
  const int hi = lane >> 4;
  const __bf16* p = base + (size_t)(row0 + l) * ldm + k0 + hi * 8;
  const v8bf lo = *(const v8bf*)p;
  const v8bf hv = *(const v8bf*)(p + 16);
  return __builtin_shufflevector(lo, hv, 0, 1, 2, 3, 4, 5, 6, 7,
                                         8, 9, 10, 11, 12, 13, 14, 15);
}

// ---------------------------------------------------------------------------
// GEMM: Y[M,Nc] = A[M,K] @ W[Nc,K]^T   (bf16 in, WMMA, fp32 accumulate/out)
// One wave owns a 16x64 tile: 1 A-frag reused across 4 B-frags per K-step.
// ---------------------------------------------------------------------------
__global__ __launch_bounds__(128)
void gemm_bf16wmma_kernel(const __bf16* __restrict__ A, const __bf16* __restrict__ W,
                          float* __restrict__ Y, int M, int K, int Nc, int apply_silu) {
  const int lane = threadIdx.x & 31;
  const int wave = threadIdx.x >> 5;
  const int wid  = blockIdx.x * 4 + wave;
  const int ngroups = Nc >> 6;          // 64 output columns per wave
  const int mt = wid / ngroups;
  const int ng = wid % ngroups;
  const int m0 = mt * 16;
  const int n0 = ng * 64;
  if (m0 >= M) return;

  v8f acc0 = {}, acc1 = {}, acc2 = {}, acc3 = {};

  for (int kc = 0; kc < K; kc += 32) {
    v16bf a  = load_frag(A, K, m0, kc, lane);
    v16bf b0 = load_frag(W, K, n0 +  0, kc, lane);
    v16bf b1 = load_frag(W, K, n0 + 16, kc, lane);
    v16bf b2 = load_frag(W, K, n0 + 32, kc, lane);
    v16bf b3 = load_frag(W, K, n0 + 48, kc, lane);
    if (kc + 32 < K) {   // hide L2 latency behind the WMMA issue stream
      __builtin_prefetch(A + (size_t)(m0 + (lane & 15)) * K + kc + 32, 0, 1);
      __builtin_prefetch(W + (size_t)(n0 + (lane & 15)) * K + kc + 32, 0, 1);
      __builtin_prefetch(W + (size_t)(n0 + 32 + (lane & 15)) * K + kc + 32, 0, 1);
    }
    acc0 = __builtin_amdgcn_wmma_f32_16x16x32_bf16(false, a, false, b0, (short)0, acc0, false, false);
    acc1 = __builtin_amdgcn_wmma_f32_16x16x32_bf16(false, a, false, b1, (short)0, acc1, false, false);
    acc2 = __builtin_amdgcn_wmma_f32_16x16x32_bf16(false, a, false, b2, (short)0, acc2, false, false);
    acc3 = __builtin_amdgcn_wmma_f32_16x16x32_bf16(false, a, false, b3, (short)0, acc3, false, false);
  }

  // C/D layout: VGPR r, lanes 0-15 -> M=r, N=lane; lanes 16-31 -> M=8+r, N=lane-16
  const int col  = lane & 15;
  const int rofs = (lane >> 4) * 8;
#pragma unroll
  for (int r = 0; r < 8; ++r) {
    const size_t row = (size_t)(m0 + rofs + r);
    float v0 = acc0[r], v1 = acc1[r], v2 = acc2[r], v3 = acc3[r];
    if (apply_silu) {
      v0 = v0 / (1.0f + __expf(-v0));
      v1 = v1 / (1.0f + __expf(-v1));
      v2 = v2 / (1.0f + __expf(-v2));
      v3 = v3 / (1.0f + __expf(-v3));
    }
    Y[row * Nc + n0 + col +  0] = v0;
    Y[row * Nc + n0 + col + 16] = v1;
    Y[row * Nc + n0 + col + 32] = v2;
    Y[row * Nc + n0 + col + 48] = v3;
  }
}

// ---------------------------------------------------------------------------
// fp32 -> bf16 weight conversion (one-shot, 2.6 MB total: lives in L2)
// ---------------------------------------------------------------------------
__global__ __launch_bounds__(256)
void cvt_bf16_kernel(const float* __restrict__ src, __bf16* __restrict__ dst, int n) {
  const int i = blockIdx.x * 256 + threadIdx.x;
  if (i < n) dst[i] = (__bf16)src[i];
}

// ---------------------------------------------------------------------------
// Fused token-shift + LoRA mixing + decay. One block per token.
// Emits xk/xv/xr/xg directly in bf16 (GEMM currency); w_eff in fp32.
// ---------------------------------------------------------------------------
__global__ __launch_bounds__(256)
void mix_kernel(const float* __restrict__ x,
                const float* __restrict__ maa_x, const float* __restrict__ maa_w,
                const float* __restrict__ maa_k, const float* __restrict__ maa_v,
                const float* __restrict__ maa_r, const float* __restrict__ maa_g,
                const float* __restrict__ w1, const float* __restrict__ w2,
                const float* __restrict__ tdecay, const float* __restrict__ dw1,
                const float* __restrict__ dw2,
                __bf16* __restrict__ xr, __bf16* __restrict__ xk,
                __bf16* __restrict__ xv, __bf16* __restrict__ xg,
                float* __restrict__ weff) {
  __shared__ float xs[CDIM];
  __shared__ float xxs[CDIM];
  __shared__ float tmp[CDIM];          // xxx, then reused for xw
  __shared__ float hs[5 * DMIXD];
  __shared__ float hds[DDECD];

  const int bt  = blockIdx.x;
  const int t   = bt % TDIM;
  const int tid = threadIdx.x;
  const float* xrow  = x + (size_t)bt * CDIM;
  const float* xprev = xrow - CDIM;

  for (int c = tid; c < CDIM; c += 256) {
    const float xc = xrow[c];
    const float xp = (t == 0) ? 0.0f : xprev[c];
    const float d  = xp - xc;
    xs[c]  = xc;
    xxs[c] = d;
    tmp[c] = xc + d * maa_x[c];        // xxx
  }
  __syncthreads();

  // h = tanh(xxx @ w1)   w1: [C, 160]
  if (tid < 5 * DMIXD) {
    float acc = 0.0f;
    for (int c = 0; c < CDIM; ++c) acc += tmp[c] * w1[c * (5 * DMIXD) + tid];
    hs[tid] = tanhf(acc);
  }
  __syncthreads();

  // five branch projections: m_f = h_f @ w2_f  (w2: [5, 32, C])
  const float* maas[5] = {maa_w, maa_k, maa_v, maa_r, maa_g};
  __bf16* outs[5] = {nullptr, xk, xv, xr, xg};
#pragma unroll
  for (int f = 0; f < 5; ++f) {
    const float* hf  = &hs[f * DMIXD];
    const float* w2f = w2 + (size_t)f * DMIXD * CDIM;
    for (int c = tid; c < CDIM; c += 256) {
      float m = 0.0f;
#pragma unroll
      for (int d = 0; d < DMIXD; ++d) m += hf[d] * w2f[d * CDIM + c];
      const float val = xs[c] + xxs[c] * (maas[f][c] + m);
      if (f == 0) tmp[c] = val;                       // xw -> LDS (tmp is dead)
      else        outs[f][(size_t)bt * CDIM + c] = (__bf16)val;
    }
  }
  __syncthreads();

  // decay LoRA: hd = tanh(xw @ dw1)   dw1: [C, 64]
  if (tid < DDECD) {
    float acc = 0.0f;
    for (int c = 0; c < CDIM; ++c) acc += tmp[c] * dw1[c * DDECD + tid];
    hds[tid] = tanhf(acc);
  }
  __syncthreads();

  // w_eff = exp(-exp(time_decay + hd @ dw2))   dw2: [64, C]
  for (int c = tid; c < CDIM; c += 256) {
    float ww = 0.0f;
#pragma unroll
    for (int j = 0; j < DDECD; ++j) ww += hds[j] * dw2[j * CDIM + c];
    const float wr = tdecay[c] + ww;
    weff[(size_t)bt * CDIM + c] = __expf(-__expf(wr));
  }
}

// ---------------------------------------------------------------------------
// WKV6 recurrence. One block per (b,h); 64 threads, thread i owns state
// column S[:, i] in 64 VGPRs. r/k/w broadcast through LDS each step.
// y[i] = sum_j r[j]*(S[j,i] + u[j]*k[j]*v[i]);  S[j,i] = w[j]*S[j,i] + k[j]*v[i]
// ---------------------------------------------------------------------------
__global__ __launch_bounds__(64)
void wkv_kernel(const float* __restrict__ r, const float* __restrict__ k,
                const float* __restrict__ v, const float* __restrict__ w,
                const float* __restrict__ u, float* __restrict__ y) {
  const int bh = blockIdx.x;
  const int b  = bh / HDIM;
  const int h  = bh % HDIM;
  const int i  = threadIdx.x;

  __shared__ float rs[NDIM], ks[NDIM], wsd[NDIM], us[NDIM];
  us[i] = u[h * NDIM + i];

  float S[NDIM];
#pragma unroll
  for (int j = 0; j < NDIM; ++j) S[j] = 0.0f;

  size_t base = (size_t)b * TDIM * CDIM + h * NDIM;
  for (int t = 0; t < TDIM; ++t) {
    __syncthreads();
    rs[i]  = r[base + i];
    ks[i]  = k[base + i];
    wsd[i] = w[base + i];
    const float vi = v[base + i];
    __syncthreads();

    float yi = 0.0f;
#pragma unroll 8
    for (int j = 0; j < NDIM; ++j) {
      const float kv = ks[j] * vi;
      yi  += rs[j] * (S[j] + us[j] * kv);
      S[j] = wsd[j] * S[j] + kv;
    }
    y[base + i] = yi;
    base += CDIM;
  }
}

// ---------------------------------------------------------------------------
// GroupNorm (H groups of N) + gate multiply. One wave per head; 64 values =
// 2 per lane, butterfly shfl_xor reduction (wave32-native). Emits bf16.
// ---------------------------------------------------------------------------
__global__ __launch_bounds__(256)
void gn_mul_kernel(const float* __restrict__ y, const float* __restrict__ g,
                   const float* __restrict__ lng, const float* __restrict__ lnb,
                   __bf16* __restrict__ yg) {
  const int bt   = blockIdx.x;
  const int head = threadIdx.x >> 5;
  const int lane = threadIdx.x & 31;
  const size_t base = (size_t)bt * CDIM + head * NDIM;

  const float v0 = y[base + lane];
  const float v1 = y[base + lane + 32];
  float s  = v0 + v1;
  float sq = v0 * v0 + v1 * v1;
#pragma unroll
  for (int m = 16; m >= 1; m >>= 1) {
    s  += __shfl_xor(s,  m, 32);
    sq += __shfl_xor(sq, m, 32);
  }
  const float mean = s * (1.0f / NDIM);
  const float var  = sq * (1.0f / NDIM) - mean * mean;
  const float rstd = rsqrtf(var + 1e-5f);

  const int c0 = head * NDIM + lane;
  const float yn0 = (v0 - mean) * rstd * lng[c0]      + lnb[c0];
  const float yn1 = (v1 - mean) * rstd * lng[c0 + 32] + lnb[c0 + 32];
  yg[base + lane]      = (__bf16)(yn0 * g[base + lane]);
  yg[base + lane + 32] = (__bf16)(yn1 * g[base + lane + 32]);
}

// ---------------------------------------------------------------------------
extern "C" void kernel_launch(void* const* d_in, const int* in_sizes, int n_in,
                              void* d_out, int out_size, void* d_ws, size_t ws_size,
                              hipStream_t stream) {
  const float* x      = (const float*)d_in[0];
  const float* maa_x  = (const float*)d_in[1];
  const float* maa_w  = (const float*)d_in[2];
  const float* maa_k  = (const float*)d_in[3];
  const float* maa_v  = (const float*)d_in[4];
  const float* maa_r  = (const float*)d_in[5];
  const float* maa_g  = (const float*)d_in[6];
  const float* w1     = (const float*)d_in[7];
  const float* w2     = (const float*)d_in[8];
  const float* tdecay = (const float*)d_in[9];
  const float* dw1    = (const float*)d_in[10];
  const float* dw2    = (const float*)d_in[11];
  const float* u      = (const float*)d_in[12];
  const float* Wr     = (const float*)d_in[13];
  const float* Wk     = (const float*)d_in[14];
  const float* Wv     = (const float*)d_in[15];
  const float* Wg     = (const float*)d_in[16];
  const float* Wo     = (const float*)d_in[17];
  const float* lng    = (const float*)d_in[18];
  const float* lnb    = (const float*)d_in[19];
  float* out = (float*)d_out;

  const size_t MC = (size_t)BDIM * TDIM * CDIM;   // 8,388,608 elements
  const size_t WN = (size_t)CDIM * CDIM;          // 262,144 elements

  char* p = (char*)d_ws;
  float* weff = (float*)p; p += MC * sizeof(float);
  float* rbuf = (float*)p; p += MC * sizeof(float);
  float* kbuf = (float*)p; p += MC * sizeof(float);
  float* vbuf = (float*)p; p += MC * sizeof(float);
  float* gbuf = (float*)p; p += MC * sizeof(float);
  float* ybuf = (float*)p; p += MC * sizeof(float);
  __bf16* xrb = (__bf16*)p; p += MC * sizeof(__bf16);
  __bf16* xkb = (__bf16*)p; p += MC * sizeof(__bf16);
  __bf16* xvb = (__bf16*)p; p += MC * sizeof(__bf16);
  __bf16* xgb = (__bf16*)p; p += MC * sizeof(__bf16);
  __bf16* ygb = (__bf16*)p; p += MC * sizeof(__bf16);
  __bf16* Wrb = (__bf16*)p; p += WN * sizeof(__bf16);
  __bf16* Wkb = (__bf16*)p; p += WN * sizeof(__bf16);
  __bf16* Wvb = (__bf16*)p; p += WN * sizeof(__bf16);
  __bf16* Wgb = (__bf16*)p; p += WN * sizeof(__bf16);
  __bf16* Wob = (__bf16*)p; p += WN * sizeof(__bf16);

  const int M = BDIM * TDIM;                       // 16384
  const int gemm_blocks = (M / 16) * (CDIM / 64) / 4;   // 8192 waves / 4
  const int cvt_blocks  = (int)((WN + 255) / 256);

  // 0) weights -> bf16 (2.6 MB total; stays resident in the 192 MB L2)
  cvt_bf16_kernel<<<cvt_blocks, 256, 0, stream>>>(Wr, Wrb, (int)WN);
  cvt_bf16_kernel<<<cvt_blocks, 256, 0, stream>>>(Wk, Wkb, (int)WN);
  cvt_bf16_kernel<<<cvt_blocks, 256, 0, stream>>>(Wv, Wvb, (int)WN);
  cvt_bf16_kernel<<<cvt_blocks, 256, 0, stream>>>(Wg, Wgb, (int)WN);
  cvt_bf16_kernel<<<cvt_blocks, 256, 0, stream>>>(Wo, Wob, (int)WN);

  // 1) token shift + mixing + decay (emits bf16 activations)
  mix_kernel<<<M, 256, 0, stream>>>(x, maa_x, maa_w, maa_k, maa_v, maa_r, maa_g,
                                    w1, w2, tdecay, dw1, dw2,
                                    xrb, xkb, xvb, xgb, weff);

  // 2) four projection GEMMs (pure bf16 WMMA path)
  gemm_bf16wmma_kernel<<<gemm_blocks, 128, 0, stream>>>(xrb, Wrb, rbuf, M, CDIM, CDIM, 0);
  gemm_bf16wmma_kernel<<<gemm_blocks, 128, 0, stream>>>(xkb, Wkb, kbuf, M, CDIM, CDIM, 0);
  gemm_bf16wmma_kernel<<<gemm_blocks, 128, 0, stream>>>(xvb, Wvb, vbuf, M, CDIM, CDIM, 0);
  gemm_bf16wmma_kernel<<<gemm_blocks, 128, 0, stream>>>(xgb, Wgb, gbuf, M, CDIM, CDIM, 1);

  // 3) serial WKV6 scan (latency-bound; VALU)
  wkv_kernel<<<BDIM * HDIM, 64, 0, stream>>>(rbuf, kbuf, vbuf, weff, u, ybuf);

  // 4) groupnorm + gate (emits bf16)
  gn_mul_kernel<<<M, 256, 0, stream>>>(ybuf, gbuf, lng, lnb, ygb);

  // 5) output projection
  gemm_bf16wmma_kernel<<<gemm_blocks, 128, 0, stream>>>(ygb, Wob, out, M, CDIM, CDIM, 0);
}
